// GraphRegressorMLP_1949915152980
// MI455X (gfx1250) — matrix operand order repaired and checked by reading, not compiled, not run on hardware
//
#include <hip/hip_runtime.h>
#include <hip/hip_bf16.h>

// ---------------------------------------------------------------------------
// GraphRegressorMLP on gfx1250 (MI455X, wave32, WMMA).
// Big GEMMs -> v_wmma_f32_16x16x32_f16 (f16 in, f32 accumulate).
// Edge-feature gather -> global_load_async_to_lds_b128 (ASYNCcnt).
// Edge kernel: 32 edges/block, 2 M-tiles per wave sharing B fragments.
// ---------------------------------------------------------------------------

typedef __attribute__((ext_vector_type(16))) _Float16 v16h;
typedef __attribute__((ext_vector_type(8)))  float    v8f;

#define NUM_GRAPHS 64
#define H 128

#define K1     272   // real K of edge layer 1 (2H + 16)
#define K1PAD  288   // K padded to multiple of 32
#define EFSTR  296   // LDS row stride (halves) for ef   (148 dw, conflict-free)
#define TLSTR  136   // LDS row stride (halves) for 128-col tiles (68 dw, conflict-free)
#define EROWS  32    // edges per block (two 16-row M-tiles)

struct alignas(16) H8 { _Float16 h[8]; };

static __device__ inline v8f wmma16x16x32(v16h a, v16h b, v8f c) {
    // (neg_a, A, neg_b, B, c_mod, C, reuse_a, reuse_b)
    return __builtin_amdgcn_wmma_f32_16x16x32_f16(false, a, false, b,
                                                  (short)0, c, false, false);
}

static __device__ inline void wait_async0() {
#if __has_builtin(__builtin_amdgcn_s_wait_asynccnt)
    __builtin_amdgcn_s_wait_asynccnt(0);
#else
    asm volatile("s_wait_asynccnt 0x0" ::: "memory");
#endif
}

// Async DMA: 16 bytes global -> LDS, per lane, tracked on ASYNCcnt.
static __device__ inline void async_copy_b128(unsigned lds_off, const void* gsrc) {
    unsigned long long ga = (unsigned long long)(size_t)gsrc;
    asm volatile("global_load_async_to_lds_b128 %0, %1, off"
                 :: "v"(lds_off), "v"(ga) : "memory");
}

// A-fragment (16x32 f16, MxK) from an LDS tile, row stride `lda` halves.
// Lanes 0-15: M=lane, halves 0-7 = K[k0..k0+7], 8-15 = K[k0+16..k0+23];
// lanes 16-31: M=lane-16, halves 0-7 = K[k0+8..], 8-15 = K[k0+24..].
static __device__ inline v16h load_a_lds(const _Float16* A, int lda, int k0, int lane) {
    int m  = lane & 15;
    int lo = (lane < 16) ? 0 : 8;
    const H8* p0 = (const H8*)(A + m * lda + k0 + lo);
    const H8* p1 = (const H8*)(A + m * lda + k0 + lo + 16);
    union { v16h v; H8 p[2]; } u;
    u.p[0] = *p0;
    u.p[1] = *p1;
    return u.v;
}

// B-fragment from pre-packed weights (packed exactly in lane order: see pack_b).
static __device__ inline v16h load_b_packed(const _Float16* P, int nsteps,
                                            int nblk, int kstep, int lane) {
    return *(const v16h*)(P + ((((size_t)nblk * nsteps) + kstep) * 32 + lane) * 16);
}

// ---------------------------------------------------------------------------
// Pack a KxN(=128) f32 weight matrix into f16 B-fragment order.
// B layout (32x16 KxN per fragment): lane holds column n0+(lane&15),
// halves i<8 = W[k0 + (lane<16?0:8) + i][n], i>=8 = W[k0 + 16 + (lane<16?0:8) + i-8][n]. Rows >= krows are zero-padded.
// ---------------------------------------------------------------------------
__global__ void pack_b_kernel(const float* __restrict__ W, _Float16* __restrict__ out,
                              int krows, int nsteps) {
    int idx = blockIdx.x * 256 + threadIdx.x;
    int total = 8 * nsteps * 32 * 16;           // 8 n-blocks of 16 cols (N = 128)
    if (idx >= total) return;
    int i     = idx & 15;
    int lane  = (idx >> 4) & 31;
    int rest  = idx >> 9;                       // nblk * nsteps + kstep
    int kstep = rest % nsteps;
    int nblk  = rest / nsteps;
    int n = nblk * 16 + (lane & 15);
    int k = kstep * 32 + ((lane < 16) ? 0 : 8) + (i & 7) + ((i & 8) << 1);
    out[idx] = (k < krows) ? (_Float16)W[(size_t)k * H + n] : (_Float16)0.f;
}

__global__ void init_kernel(float* __restrict__ sums, float* __restrict__ counts) {
    int tid = threadIdx.x;
    for (int i = tid; i < NUM_GRAPHS * H; i += 256) sums[i] = 0.f;
    if (tid < NUM_GRAPHS) counts[tid] = 0.f;
}

// ---------------------------------------------------------------------------
// Node MLP: h = relu(x @ node_w + node_b), output stored as f16.
// 16 nodes per block; 8 waves, wave w owns output columns [16w, 16w+16).
// ---------------------------------------------------------------------------
__global__ void __launch_bounds__(256)
node_mlp_kernel(const float* __restrict__ x, const _Float16* __restrict__ Wp,
                const float* __restrict__ bias, _Float16* __restrict__ h_out, int N) {
    __shared__ _Float16 xl[16 * TLSTR];
    int tid = threadIdx.x;
    int nodeBase = blockIdx.x * 16;
    for (int idx = tid; idx < 16 * H; idx += 256) {
        int r = idx >> 7, c = idx & 127;
        int row = nodeBase + r;
        float v = (row < N) ? x[(size_t)row * H + c] : 0.f;
        xl[r * TLSTR + c] = (_Float16)v;
    }
    __syncthreads();

    int lane = tid & 31;
    int w    = tid >> 5;
    v8f acc = {};
#pragma unroll
    for (int ks = 0; ks < 4; ++ks) {            // K = 128
        v16h a = load_a_lds(xl, TLSTR, ks * 32, lane);
        v16h b = load_b_packed(Wp, 4, w, ks, lane);
        acc = wmma16x16x32(a, b, acc);
    }
    __syncthreads();                            // all waves done reading xl

    int n  = w * 16 + (lane & 15);
    float bv = bias[n];
    int mbase = (lane < 16) ? 0 : 8;
#pragma unroll
    for (int r = 0; r < 8; ++r) {
        float v = acc[r] + bv;
        xl[(mbase + r) * TLSTR + n] = (_Float16)(v > 0.f ? v : 0.f);
    }
    __syncthreads();

    // Coalesced write-back: one global_store_b128 per thread.
    {
        int r  = tid >> 4;                      // 0..15
        int c8 = tid & 15;                      // 8-half chunk
        int row = nodeBase + r;
        if (row < N)
            *(H8*)(h_out + (size_t)row * H + c8 * 8) =
                *(const H8*)(xl + r * TLSTR + c8 * 8);
    }
}

// ---------------------------------------------------------------------------
// Edge-per-graph histogram for the segment mean denominator.
// ---------------------------------------------------------------------------
__global__ void __launch_bounds__(256)
count_kernel(const int* __restrict__ ei, const int* __restrict__ batch,
             float* __restrict__ counts, int E) {
    __shared__ float loc[NUM_GRAPHS];
    int tid = threadIdx.x;
    if (tid < NUM_GRAPHS) loc[tid] = 0.f;
    __syncthreads();
    int e = blockIdx.x * 256 + tid;
    if (e < E) atomicAdd(&loc[batch[ei[e]]], 1.f);
    __syncthreads();
    if (tid < NUM_GRAPHS) {
        float v = loc[tid];
        if (v != 0.f) atomicAdd(&counts[tid], v);
    }
}

// ---------------------------------------------------------------------------
// Fused edge MLP + segment-sum. 32 edges per block, 2 M-tiles per wave.
//   ef  = concat(h[parent], h[child], edge_attr)  -> [32 x 288] f16
//         (h rows async-DMA'd global->LDS, ASYNCcnt)
//   t   = relu(ef @ W1 + b1)                      -> [32 x 128] f16 in LDS
//   e   = t @ W2 + b2                             -> atomicAdd into sums[graph]
// Each wave owns 16 output cols; both M-tiles share every B fragment.
// ---------------------------------------------------------------------------
__global__ void __launch_bounds__(256)
edge_mlp_kernel(const int* __restrict__ ei, const float* __restrict__ eattr,
                const int* __restrict__ batch, const _Float16* __restrict__ hfeat,
                const _Float16* __restrict__ W1p, const float* __restrict__ b1,
                const _Float16* __restrict__ W2p, const float* __restrict__ b2,
                float* __restrict__ sums, int E) {
    __shared__ _Float16 ef[EROWS * EFSTR];
    __shared__ _Float16 tl[EROWS * TLSTR];
    __shared__ int sp[EROWS], sc[EROWS], sg[EROWS];

    int tid  = threadIdx.x;
    int base = blockIdx.x * EROWS;

    if (tid < EROWS) {
        int e = base + tid;
        int p = 0, c = 0;
        if (e < E) { p = ei[e]; c = ei[E + e]; }
        sp[tid] = p; sc[tid] = c; sg[tid] = batch[p];
    }
    __syncthreads();

    // Gather h[parent] (cols 0-127) and h[child] (cols 128-255) as raw f16
    // bytes: 1024 x 16B async DMA chunks, no VGPR round-trip.
    {
        unsigned efbase = (unsigned)(size_t)(void*)ef;  // wave-relative LDS offset
#pragma unroll
        for (int it = 0; it < 4; ++it) {
            int q  = tid + it * 256;                    // 0..1023
            int r  = q >> 5;                            // edge row 0..31
            int c8 = q & 31;                            // 8-half chunk 0..31
            int node = (c8 < 16) ? sp[r] : sc[r];
            const _Float16* src = hfeat + (size_t)node * H + (size_t)(c8 & 15) * 8;
            unsigned lds = efbase + (unsigned)(r * EFSTR + c8 * 8) * 2u;
            async_copy_b128(lds, src);
        }
    }
    // edge_attr tail (f32 -> f16) + zero pad, cols 256..287.
    for (int idx = tid; idx < EROWS * 32; idx += 256) {
        int r = idx >> 5;
        int c = 256 + (idx & 31);
        int e = base + r;
        _Float16 v = (_Float16)0.f;
        if (e < E && c < K1) v = (_Float16)eattr[(size_t)e * 16 + (c - 256)];
        ef[r * EFSTR + c] = v;
    }
    wait_async0();
    __syncthreads();

    int lane  = tid & 31;
    int w     = tid >> 5;
    int n     = w * 16 + (lane & 15);
    int mbase = (lane < 16) ? 0 : 8;

    // Layer 1: [32 x 288] @ [288 x 128]; B fragment shared by both M-tiles.
    v8f acc0 = {}, acc1 = {};
#pragma unroll
    for (int ks = 0; ks < 9; ++ks) {
        v16h b  = load_b_packed(W1p, 9, w, ks, lane);
        v16h a0 = load_a_lds(ef,              EFSTR, ks * 32, lane);
        v16h a1 = load_a_lds(ef + 16 * EFSTR, EFSTR, ks * 32, lane);
        acc0 = wmma16x16x32(a0, b, acc0);
        acc1 = wmma16x16x32(a1, b, acc1);
    }
    {
        float bv = b1[n];
#pragma unroll
        for (int r = 0; r < 8; ++r) {
            float v0 = acc0[r] + bv;
            float v1 = acc1[r] + bv;
            tl[(mbase + r) * TLSTR + n]        = (_Float16)(v0 > 0.f ? v0 : 0.f);
            tl[(16 + mbase + r) * TLSTR + n]   = (_Float16)(v1 > 0.f ? v1 : 0.f);
        }
    }
    __syncthreads();

    // Layer 2: [32 x 128] @ [128 x 128]
    v8f d0 = {}, d1 = {};
#pragma unroll
    for (int ks = 0; ks < 4; ++ks) {
        v16h b  = load_b_packed(W2p, 4, w, ks, lane);
        v16h a0 = load_a_lds(tl,              TLSTR, ks * 32, lane);
        v16h a1 = load_a_lds(tl + 16 * TLSTR, TLSTR, ks * 32, lane);
        d0 = wmma16x16x32(a0, b, d0);
        d1 = wmma16x16x32(a1, b, d1);
    }
    float bv2 = b2[n];
    if (base + EROWS <= E) {                     // uniform fast path (always, E%32==0)
#pragma unroll
        for (int r = 0; r < 8; ++r) {
            atomicAdd(&sums[sg[mbase + r] * H + n],      d0[r] + bv2);
            atomicAdd(&sums[sg[16 + mbase + r] * H + n], d1[r] + bv2);
        }
    } else {
#pragma unroll
        for (int r = 0; r < 8; ++r) {
            int m0 = mbase + r, m1 = 16 + mbase + r;
            if (base + m0 < E) atomicAdd(&sums[sg[m0] * H + n], d0[r] + bv2);
            if (base + m1 < E) atomicAdd(&sums[sg[m1] * H + n], d1[r] + bv2);
        }
    }
}

// ---------------------------------------------------------------------------
// Mean pool + regression head (tiny: 64 graphs). Single block.
// ---------------------------------------------------------------------------
__global__ void __launch_bounds__(128)
finalize_kernel(const float* __restrict__ sums, const float* __restrict__ counts,
                const float* __restrict__ w1, const float* __restrict__ b1,
                const float* __restrict__ w2, const float* __restrict__ b2,
                float* __restrict__ out) {
    __shared__ float emb[NUM_GRAPHS * H];
    __shared__ float t1[NUM_GRAPHS * 64];
    int tid = threadIdx.x;
    for (int i = tid; i < NUM_GRAPHS * H; i += 128) {
        int g = i >> 7;
        emb[i] = sums[i] / fmaxf(counts[g], 1.f);
    }
    __syncthreads();
    for (int i = tid; i < NUM_GRAPHS * 64; i += 128) {
        int g = i >> 6, j = i & 63;
        float s = b1[j];
        const float* er = &emb[g * H];
        for (int k = 0; k < H; ++k) s += er[k] * w1[k * 64 + j];
        t1[i] = s > 0.f ? s : 0.f;
    }
    __syncthreads();
    if (tid < NUM_GRAPHS) {
        float s = b2[0];
        const float* tr = &t1[tid * 64];
        for (int k = 0; k < 64; ++k) s += tr[k] * w2[k];
        out[tid] = s;
    }
}

// ---------------------------------------------------------------------------
extern "C" void kernel_launch(void* const* d_in, const int* in_sizes, int n_in,
                              void* d_out, int out_size, void* d_ws, size_t ws_size,
                              hipStream_t stream) {
    const float* x          = (const float*)d_in[0];
    const int*   edge_index = (const int*)  d_in[1];
    const float* edge_attr  = (const float*)d_in[2];
    const int*   batch      = (const int*)  d_in[3];
    const float* node_w     = (const float*)d_in[4];
    const float* node_b     = (const float*)d_in[5];
    const float* e_w1       = (const float*)d_in[6];
    const float* e_b1       = (const float*)d_in[7];
    const float* e_w2       = (const float*)d_in[8];
    const float* e_b2       = (const float*)d_in[9];
    const float* h_w1       = (const float*)d_in[10];
    const float* h_b1       = (const float*)d_in[11];
    const float* h_w2       = (const float*)d_in[12];
    const float* h_b2       = (const float*)d_in[13];

    const int N = in_sizes[0] / H;      // 50000
    const int E = in_sizes[1] / 2;      // 800000

    // Workspace layout (all 256B aligned).
    char* ws = (char*)d_ws;
    size_t off = 0;
    auto alloc = [&](size_t bytes) {
        char* p = ws + off;
        off = (off + bytes + 255) & ~(size_t)255;
        return p;
    };
    _Float16* h_half = (_Float16*)alloc((size_t)N * H * sizeof(_Float16));
    float*    sums   = (float*)   alloc(NUM_GRAPHS * H * sizeof(float));
    float*    counts = (float*)   alloc(NUM_GRAPHS * sizeof(float));
    _Float16* pwN    = (_Float16*)alloc((size_t)8 * 4 * 512 * sizeof(_Float16)); // node_w
    _Float16* pw1    = (_Float16*)alloc((size_t)8 * 9 * 512 * sizeof(_Float16)); // e_w1
    _Float16* pw2    = (_Float16*)alloc((size_t)8 * 4 * 512 * sizeof(_Float16)); // e_w2

    // Repack weights into f16 B-fragment order (cheap; done every call).
    pack_b_kernel<<<(8 * 4 * 512 + 255) / 256, 256, 0, stream>>>(node_w, pwN, 128, 4);
    pack_b_kernel<<<(8 * 9 * 512 + 255) / 256, 256, 0, stream>>>(e_w1, pw1, K1, 9);
    pack_b_kernel<<<(8 * 4 * 512 + 255) / 256, 256, 0, stream>>>(e_w2, pw2, 128, 4);

    init_kernel<<<1, 256, 0, stream>>>(sums, counts);

    node_mlp_kernel<<<(N + 15) / 16, 256, 0, stream>>>(x, pwN, node_b, h_half, N);

    count_kernel<<<(E + 255) / 256, 256, 0, stream>>>(edge_index, batch, counts, E);

    edge_mlp_kernel<<<(E + EROWS - 1) / EROWS, 256, 0, stream>>>(
        edge_index, edge_attr, batch, h_half, pw1, e_b1, pw2, e_b2, sums, E);

    finalize_kernel<<<1, 128, 0, stream>>>(sums, counts, h_w1, h_b1, h_w2, h_b2,
                                           (float*)d_out);
}
